// AttentionHead_22282290332408
// MI455X (gfx1250) — matrix operand order repaired and checked by reading, not compiled, not run on hardware
//
#include <hip/hip_runtime.h>
#include <hip/hip_bf16.h>

typedef __attribute__((ext_vector_type(16))) _Float16 v16h;
typedef __attribute__((ext_vector_type(8)))  float    v8f;
typedef __attribute__((ext_vector_type(4)))  int      v4i;

#define BLOCK_M   64
#define BLOCK_N   64
#define NWAVES    4
#define LSTRIDE   72          // 72 halfwords = 144B row stride: 16B aligned, conflict-free
#define NEG_INF   (-__builtin_inff())

#define AS1 __attribute__((address_space(1)))
#define AS3 __attribute__((address_space(3)))

#if __has_builtin(__builtin_amdgcn_global_load_async_to_lds_b128)
#define HAVE_ASYNC 1
// signature (from hipcc diagnostic): (v4i AS1*, v4i AS3*, imm offset, imm cpol)
#define ASYNC_CP_B128(gsrc, ldst)                                              \
  __builtin_amdgcn_global_load_async_to_lds_b128(                              \
      (AS1 v4i*)(const void*)(gsrc), (AS3 v4i*)(void*)(ldst), 0, 0)
#else
#define HAVE_ASYNC 0
#endif

#if __has_builtin(__builtin_amdgcn_s_wait_asynccnt)
#define WAIT_ASYNC(n) __builtin_amdgcn_s_wait_asynccnt(n)
#else
#define WAIT_ASYNC(n) asm volatile("s_wait_asynccnt %0" ::"n"(n) : "memory")
#endif

// A-operand layout (ISA 7.12.2, 16-bit A 16x32):
//   lane half h: elems 0..7  = K[off + h*8 + i], elems 8..15 = K[off + 16 + h*8 + i]
__device__ __forceinline__ v16h loadA_f32(const float* row, int off, int h) {
  v16h a;
#pragma unroll
  for (int i = 0; i < 8; ++i) {
    a[i]     = (_Float16)row[off + h * 8 + i];
    a[8 + i] = (_Float16)row[off + 16 + h * 8 + i];
  }
  return a;
}

__device__ __forceinline__ v16h loadA_lds(const _Float16* row, int off, int h) {
  v16h a;
#pragma unroll
  for (int i = 0; i < 8; ++i) {
    a[i]     = row[off + h * 8 + i];
    a[8 + i] = row[off + 16 + h * 8 + i];
  }
  return a;
}

// B-operand layout (dense 32x16, 16-bit; same K order as A):
//   lane half h: elems 0..7  = K[off + h*8 + i], elems 8..15 = K[off + 16 + h*8 + i]
__device__ __forceinline__ v16h loadB_lds(const _Float16* row, int off, int h) {
  v16h b;
#pragma unroll
  for (int i = 0; i < 8; ++i) {
    b[i]     = row[off + h * 8 + i];
    b[8 + i] = row[off + 16 + h * 8 + i];
  }
  return b;
}

__device__ __forceinline__ v8f wmma_f16(v16h a, v16h b, v8f c) {
  return __builtin_amdgcn_wmma_f32_16x16x32_f16(false, a, false, b, (short)0, c,
                                                false, false);
}

// ---------------------------------------------------------------------------
// Pre-pass: K (f32,row-major) -> Kh (f16,row-major); V (f32) -> Vt (f16, [64][S])
// One 64-row tile per block; LDS transpose keeps both sides coalesced.
// ---------------------------------------------------------------------------
__global__ __launch_bounds__(256) void cvt_kernel(const float* __restrict__ k,
                                                  const float* __restrict__ v,
                                                  _Float16* __restrict__ kh,
                                                  _Float16* __restrict__ vt, int S) {
  __shared__ _Float16 t[64][LSTRIDE];
  const int tid = threadIdx.x;
  const int s0  = blockIdx.x * 64;
#pragma unroll 4
  for (int e = tid; e < 64 * 64; e += 256) {
    const int row = e >> 6;   // s within tile
    const int col = e & 63;   // dk
    const size_t g = (size_t)(s0 + row) * 64 + col;
    kh[g] = (_Float16)k[g];
    t[col][row] = (_Float16)v[g];   // transpose V through LDS
  }
  __syncthreads();
#pragma unroll 4
  for (int e = tid; e < 64 * 64; e += 256) {
    const int d = e >> 6;
    const int j = e & 63;
    vt[(size_t)d * S + s0 + j] = t[d][j];
  }
}

// ---------------------------------------------------------------------------
// Main flash-attention kernel: f16 WMMA, fp32 softmax, async double-buffered
// K/V tile staging.
// ---------------------------------------------------------------------------
__global__ __launch_bounds__(128) void fa_causal_kernel(
    const float* __restrict__ q, const _Float16* __restrict__ kh,
    const _Float16* __restrict__ vt, float* __restrict__ out, int S) {
  __shared__ _Float16 Ks[2][BLOCK_N][LSTRIDE];       // K tile, row-major [kv][dk]
  __shared__ _Float16 Vs[2][64][LSTRIDE];            // V tile, transposed [dk][kv]
  __shared__ _Float16 Ps[NWAVES][16][LSTRIDE];       // per-wave P scratch [row][kv]

  const int tid  = threadIdx.x;
  const int wave = tid >> 5;
  const int lane = tid & 31;
  const int h    = lane >> 4;
  const int ln   = lane & 15;

  const int R0   = blockIdx.x * BLOCK_M;
  const int wq   = R0 + wave * 16;
  const int qrow = wq + ln;

  // stage one 64-kv tile (8 async ops per wave: 4 K + 4 V)
  auto issue_tile = [&](int jBase, int buf) {
#if HAVE_ASYNC
#pragma unroll
    for (int it = 0; it < 4; ++it) {
      const int e   = it * 128 + tid;     // 512 chunk-units of 16B
      const int row = e >> 3;
      const int ch  = (e & 7) * 8;        // halfword offset
      ASYNC_CP_B128(kh + (size_t)(jBase + row) * 64 + ch, &Ks[buf][row][ch]);
    }
#pragma unroll
    for (int it = 0; it < 4; ++it) {
      const int e   = it * 128 + tid;
      const int row = e >> 3;             // dk row of Vt
      const int ch  = (e & 7) * 8;
      ASYNC_CP_B128(vt + (size_t)row * S + jBase + ch, &Vs[buf][row][ch]);
    }
#else
#pragma unroll
    for (int it = 0; it < 4; ++it) {
      const int e   = it * 128 + tid;
      const int row = e >> 3;
      const int ch  = (e & 7) * 8;
      *(uint4*)(&Ks[buf][row][ch]) =
          *(const uint4*)(kh + (size_t)(jBase + row) * 64 + ch);
      *(uint4*)(&Vs[buf][row][ch]) =
          *(const uint4*)(vt + (size_t)row * S + jBase + ch);
    }
#endif
  };

  // Q A-operands for this wave's 16 rows (one-time f32->f16)
  const float* qr = q + (size_t)qrow * 64;
  const v16h a0 = loadA_f32(qr, 0, h);
  const v16h a1 = loadA_f32(qr, 32, h);

  float mrow[8], lsum[8];
  v8f o[4];
#pragma unroll
  for (int r = 0; r < 8; ++r) { mrow[r] = NEG_INF; lsum[r] = 0.0f; }
#pragma unroll
  for (int d = 0; d < 4; ++d)
#pragma unroll
    for (int e = 0; e < 8; ++e) o[d][e] = 0.0f;

  const int nT = (R0 + BLOCK_M) / BLOCK_N;
  issue_tile(0, 0);

  for (int jt = 0; jt < nT; ++jt) {
    const int jBase = jt * BLOCK_N;
    const int cur   = jt & 1;

    __syncthreads();   // all waves done reading buf[1-cur] (tile jt-1)
    if (jt + 1 < nT) {
      issue_tile((jt + 1) * BLOCK_N, 1 - cur);  // overlap copy of jt+1
      if (jt + 2 < nT) {                        // pull jt+2 toward L2
        __builtin_prefetch(kh + (size_t)(jBase + 128) * 64, 0, 1);
        __builtin_prefetch(vt + (size_t)(jBase + 128), 0, 1);
      }
      WAIT_ASYNC(8);   // in-order: tile jt resident, jt+1 still in flight
    } else {
      WAIT_ASYNC(0);
    }
    __syncthreads();   // tile jt visible to all waves

    if (jBase > wq + 15) continue;  // entirely above causal diagonal

    // ---- S = Q * K^T : 4 subtiles of 16x16 ----
    v8f s[4];
#pragma unroll
    for (int n = 0; n < 4; ++n) {
      const _Float16* kr = &Ks[cur][n * 16 + ln][0];
      const v16h b0 = loadB_lds(kr, 0, h);
      const v16h b1 = loadB_lds(kr, 32, h);
      v8f c;
#pragma unroll
      for (int e = 0; e < 8; ++e) c[e] = 0.0f;
      c = wmma_f16(a0, b0, c);
      c = wmma_f16(a1, b1, c);
      s[n] = c;
    }

    // ---- online softmax (fp32), P (f16) -> per-wave LDS ----
    const float scale = 1.0f / 64.0f;
#pragma unroll
    for (int r = 0; r < 8; ++r) {
      const int R = wq + r + 8 * h;
      float mx = NEG_INF;
#pragma unroll
      for (int n = 0; n < 4; ++n) {
        const int C = jBase + n * 16 + ln;
        const float val = (R >= C) ? s[n][r] * scale : NEG_INF;
        s[n][r] = val;
        mx = fmaxf(mx, val);
      }
#pragma unroll
      for (int off = 1; off < 16; off <<= 1)
        mx = fmaxf(mx, __shfl_xor(mx, off, 32));

      const float mnew  = fmaxf(mrow[r], mx);
      const float alpha = __expf(mrow[r] - mnew);

      float sum = 0.0f;
#pragma unroll
      for (int n = 0; n < 4; ++n) {
        const float p = __expf(s[n][r] - mnew);
        sum += p;
        Ps[wave][r + 8 * h][n * 16 + ln] = (_Float16)p;
      }
#pragma unroll
      for (int off = 1; off < 16; off <<= 1)
        sum += __shfl_xor(sum, off, 32);

      lsum[r] = lsum[r] * alpha + sum;
      mrow[r] = mnew;
#pragma unroll
      for (int d = 0; d < 4; ++d) o[d][r] *= alpha;
    }

    // ---- O += P * V (same-wave LDS ops are in-order) ----
    const _Float16* pr = &Ps[wave][ln][0];
    const v16h p0 = loadA_lds(pr, 0, h);
    const v16h p1 = loadA_lds(pr, 32, h);
#pragma unroll
    for (int d = 0; d < 4; ++d) {
      const _Float16* vr = &Vs[cur][d * 16 + ln][0];
      const v16h vb0 = loadB_lds(vr, 0, h);
      const v16h vb1 = loadB_lds(vr, 32, h);
      o[d] = wmma_f16(p0, vb0, o[d]);
      o[d] = wmma_f16(p1, vb1, o[d]);
    }
  }

  // ---- epilogue: O / l ----
#pragma unroll
  for (int r = 0; r < 8; ++r) {
    const int R = wq + r + 8 * h;
    const float inv = 1.0f / lsum[r];
#pragma unroll
    for (int d = 0; d < 4; ++d)
      out[(size_t)R * 64 + d * 16 + ln] = o[d][r] * inv;
  }
  (void)S;
}

extern "C" void kernel_launch(void* const* d_in, const int* in_sizes, int n_in,
                              void* d_out, int out_size, void* d_ws, size_t ws_size,
                              hipStream_t stream) {
  const float* q = (const float*)d_in[0];
  const float* k = (const float*)d_in[1];
  const float* v = (const float*)d_in[2];
  float* out = (float*)d_out;

  const int S = in_sizes[0] / 64;            // 8192
  _Float16* kh = (_Float16*)d_ws;            // f16 K, row-major [S][64]
  _Float16* vt = kh + (size_t)S * 64;        // f16 V^T, [64][S]

  cvt_kernel<<<S / 64, 256, 0, stream>>>(k, v, kh, vt, S);
  fa_causal_kernel<<<S / BLOCK_M, 128, 0, stream>>>(q, kh, vt, out, S);
  (void)n_in; (void)out_size; (void)ws_size;
}
